// PaiNN_19061064860369
// MI455X (gfx1250) — matrix operand order repaired and checked by reading, not compiled, not run on hardware
//
#include <hip/hip_runtime.h>
#include <hip/hip_bf16.h>

#define NN   40000
#define EE   400000
#define HH   128
#define H2   256
#define H3   384
#define GG   50
#define GP   64
#define LL   3
#define NGR  256
#define CUT  6.0f

typedef __attribute__((ext_vector_type(16))) __bf16 bf16x16;
typedef __attribute__((ext_vector_type(8)))  float  f32x8;
typedef __attribute__((ext_vector_type(4)))  unsigned int u32x4;
typedef __attribute__((ext_vector_type(8)))  int    i32x8;
typedef __attribute__((ext_vector_type(4)))  int    i32x4;

// ---------------- bf16 helpers ----------------
__device__ __forceinline__ unsigned short f2bf(float f) {
  unsigned int u = __float_as_uint(f);
  unsigned int r = u + 0x7FFFu + ((u >> 16) & 1u);   // round-to-nearest-even
  return (unsigned short)(r >> 16);
}

// A fragment (16x32 bf16, CDNA5 layout): lane M = lane&15, half = lane>>4;
// VGPR 0..3 hold K = kb+half*8+{0..7} (pairs), VGPR 4..7 hold K = kb+16+half*8+{0..7}.
// Per lane: two contiguous 8-float runs from the f32 activation row, converted to bf16.
__device__ __forceinline__ bf16x16 load_a_frag(const float* __restrict__ A1, int K1,
                                               const float* __restrict__ A2, int K2,
                                               int row, int kb, int lane) {
  const int half = lane >> 4;
  union { unsigned short u[16]; bf16x16 v; } out;
  int kk[2] = { kb + half * 8, kb + 16 + half * 8 };
  #pragma unroll
  for (int c = 0; c < 2; ++c) {
    int k = kk[c];
    const float* p;
    if (k < K1) p = A1 + (size_t)row * K1 + k;
    else        p = A2 + (size_t)row * K2 + (k - K1);
    float4 x0 = *(const float4*)(p);
    float4 x1 = *(const float4*)(p + 4);
    out.u[c*8+0] = f2bf(x0.x); out.u[c*8+1] = f2bf(x0.y);
    out.u[c*8+2] = f2bf(x0.z); out.u[c*8+3] = f2bf(x0.w);
    out.u[c*8+4] = f2bf(x1.x); out.u[c*8+5] = f2bf(x1.y);
    out.u[c*8+6] = f2bf(x1.z); out.u[c*8+7] = f2bf(x1.w);
  }
  return out.v;
}

// B fragment (32x16 bf16): lane N = colbase+(lane&15); elems 0..7 = K kb+half*8+{0..7}, 8..15 = K kb+16+half*8+{0..7}.
// Weights pre-packed as [(K/16), N, 16] bf16 -> two 16B loads per lane.
__device__ __forceinline__ bf16x16 load_b_frag(const unsigned short* __restrict__ Bp,
                                               int Nld, int kb, int colbase, int lane) {
  const int half = lane >> 4;
  const int n = colbase + (lane & 15);
  const unsigned short* p = Bp + ((size_t)(kb >> 4) * Nld + n) * 16 + half * 8;
  union { uint4 q[2]; bf16x16 v; } out;
  out.q[0] = *(const uint4*)(p);
  out.q[1] = *(const uint4*)(p + (size_t)Nld * 16);
  return out.v;
}

__device__ __forceinline__ f32x8 wmma_bf16(bf16x16 a, bf16x16 b, f32x8 c) {
  return __builtin_amdgcn_wmma_f32_16x16x32_bf16(false, a, false, b, (short)0, c, false, false);
}

// ---------------- generic bf16 WMMA GEMM: out = relu?(A@B + bias) ----------------
__global__ void __launch_bounds__(256)
gemm_bf16_kernel(const float* __restrict__ A1, int K1,
                 const float* __restrict__ A2, int K2,
                 const unsigned short* __restrict__ Bp,
                 const float* __restrict__ bias,
                 float* __restrict__ out, int M, int Nld, int relu) {
  const int lane = threadIdx.x & 31;
  const int wv   = threadIdx.x >> 5;
  const int m0 = (blockIdx.x * 8 + wv) * 16;
  if (m0 >= M) return;
  const int nb = blockIdx.y * 64;
  const int Ktot = K1 + K2;
  const int row = m0 + (lane & 15);
  f32x8 acc[4] = {};
  for (int kb = 0; kb < Ktot; kb += 32) {
    bf16x16 a = load_a_frag(A1, K1, A2, K2, row, kb, lane);
    #pragma unroll
    for (int t = 0; t < 4; ++t) {
      bf16x16 b = load_b_frag(Bp, Nld, kb, nb + t * 16, lane);
      acc[t] = wmma_bf16(a, b, acc[t]);
    }
  }
  const int half = lane >> 4, l15 = lane & 15;
  #pragma unroll
  for (int t = 0; t < 4; ++t) {
    const int col = nb + t * 16 + l15;
    const float bv = bias[col];                      // always valid (zero buffer if none)
    #pragma unroll
    for (int r = 0; r < 8; ++r) {
      const int rr = m0 + half * 8 + r;
      float x = acc[t][r] + bv;
      if (relu) x = fmaxf(x, 0.0f);
      out[(size_t)rr * Nld + col] = x;
    }
  }
}

// ---------------- fused filter GEMM + gate + scatter (TDM-staged A) ----------------
// A tile (16 edges x 64 f32 = contiguous 4KB of rbf) is DMA'd into LDS by the
// Tensor Data Mover with 4-dword padding every 64 dwords (row stride 68 -> the 16
// A-rows map to distinct LDS banks; each ds_load_b128 is conflict-free).
__global__ void __launch_bounds__(256)
k_filt_gate_scatter(const float* __restrict__ rbf,
                    const unsigned short* __restrict__ Bp,
                    const float* __restrict__ bias,
                    const float* __restrict__ sall,
                    const float* __restrict__ vcopy,
                    const float* __restrict__ dvn,
                    const int* __restrict__ src,
                    const int* __restrict__ dst,
                    float* __restrict__ s,
                    float* __restrict__ v) {
  __shared__ float lds_a[8][16 * 68];                // 34 KB / block
  const int lane = threadIdx.x & 31;
  const int wv   = threadIdx.x >> 5;
  const int e0 = (blockIdx.x * 8 + wv) * 16;
  const int by = blockIdx.y;

  // --- issue per-wave TDM: rbf[e0:e0+16, 0:64] -> lds_a[wv] (padded) ---
  {
    const unsigned lds_off = (unsigned)(unsigned long long)(size_t)&lds_a[wv][0];
    const unsigned long long ga = (unsigned long long)(size_t)(rbf + (size_t)e0 * GP);
    u32x4 g0;
    g0[0] = 1u;                                        // count = 1 descriptor
    g0[1] = lds_off;                                   // LDS byte address
    g0[2] = (unsigned)(ga & 0xFFFFFFFFu);              // global_addr[31:0]
    g0[3] = (unsigned)((ga >> 32) & 0x01FFFFFFu) | (2u << 30);  // addr[56:32] | type=2
    i32x8 g1;
    g1[0] = (int)((2u << 16)    // data_size = 4 bytes
                | (1u << 20)    // pad_enable
                | (5u << 22)    // pad_interval: every 64 dwords
                | (3u << 25));  // pad_amount: 4 dwords
    g1[1] = (int)(1024u << 16); // tensor_dim0 = 1024 (lo16)
    g1[2] = (int)(1u << 16);    // tensor_dim0 hi = 0 | tensor_dim1 = 1 (lo16)
    g1[3] = (int)(1024u << 16); // tensor_dim1 hi = 0 | tile_dim0 = 1024
    g1[4] = 1;                  // tile_dim1 = 1, tile_dim2 = 0
    g1[5] = 1024;               // tensor_dim0_stride (lo32)
    g1[6] = 0;                  // stride0 hi | stride1 lo16
    g1[7] = 0;
    i32x4 gz4 = {0, 0, 0, 0};
    i32x8 gz8 = {0, 0, 0, 0, 0, 0, 0, 0};
    __builtin_amdgcn_tensor_load_to_lds(g0, g1, gz4, gz4, gz8, 0);
  }

  int cb[4];
  if (by < 4) {
    cb[0] = HH + 32 * by;     cb[1] = cb[0] + 16;     // g1 columns
    cb[2] = 2 * HH + 32 * by; cb[3] = cb[2] + 16;     // matching g2 columns
  } else {
    int b0 = (by - 4) * 64;                            // g0 columns
    cb[0] = b0; cb[1] = b0 + 16; cb[2] = b0 + 32; cb[3] = b0 + 48;
  }

  const int half = lane >> 4, l15 = lane & 15;
  const float* ap = &lds_a[wv][l15 * 68];

  __builtin_amdgcn_s_wait_tensorcnt(0);               // DMA complete before LDS reads

  f32x8 acc[4] = {};
  for (int kb = 0; kb < GP; kb += 32) {
    union { unsigned short u[16]; bf16x16 v; } fa;
    #pragma unroll
    for (int c = 0; c < 2; ++c) {
      const int k = kb + (c ? 16 : 0) + half * 8;
      float4 x0 = *(const float4*)(ap + k);
      float4 x1 = *(const float4*)(ap + k + 4);
      fa.u[c*8+0] = f2bf(x0.x); fa.u[c*8+1] = f2bf(x0.y);
      fa.u[c*8+2] = f2bf(x0.z); fa.u[c*8+3] = f2bf(x0.w);
      fa.u[c*8+4] = f2bf(x1.x); fa.u[c*8+5] = f2bf(x1.y);
      fa.u[c*8+6] = f2bf(x1.z); fa.u[c*8+7] = f2bf(x1.w);
    }
    #pragma unroll
    for (int t = 0; t < 4; ++t)
      acc[t] = wmma_bf16(fa.v, load_b_frag(Bp, H3, kb, cb[t], lane), acc[t]);
  }

  #pragma unroll
  for (int r = 0; r < 8; ++r) {
    const int e = e0 + half * 8 + r;
    const int se = src[e], de = dst[e];
    if (by < 4) {
      const float d0 = dvn[e * 3 + 0], d1 = dvn[e * 3 + 1], d2 = dvn[e * 3 + 2];
      #pragma unroll
      for (int t = 0; t < 2; ++t) {
        const int c1 = cb[t] + l15;       // g1 col (128..255)
        const int c2 = cb[t + 2] + l15;   // g2 col (256..383)
        const int hc = c1 - HH;
        const float g1 = (acc[t][r]     + bias[c1]) * sall[(size_t)se * H3 + c1];
        const float g2 = (acc[t + 2][r] + bias[c2]) * sall[(size_t)se * H3 + c2];
        atomicAdd(&v[((size_t)de * 3 + 0) * HH + hc], vcopy[((size_t)se * 3 + 0) * HH + hc] * g1 + d0 * g2);
        atomicAdd(&v[((size_t)de * 3 + 1) * HH + hc], vcopy[((size_t)se * 3 + 1) * HH + hc] * g1 + d1 * g2);
        atomicAdd(&v[((size_t)de * 3 + 2) * HH + hc], vcopy[((size_t)se * 3 + 2) * HH + hc] * g1 + d2 * g2);
      }
    } else {
      #pragma unroll
      for (int t = 0; t < 4; ++t) {
        const int c = cb[t] + l15;        // g0 col (0..127)
        const float g0 = (acc[t][r] + bias[c]) * sall[(size_t)se * H3 + c];
        atomicAdd(&s[(size_t)de * HH + c], g0);
      }
    }
  }
}

// ---------------- weight repack: f32 [K,N] -> bf16 packed [(Kpad/16), N, 16] ----------------
__global__ void pack_b_kernel(const float* __restrict__ srcw, unsigned short* __restrict__ dstw,
                              int Ksrc, int Kpad, int Nc) {
  int idx = blockIdx.x * 256 + threadIdx.x;
  if (idx >= Kpad * Nc) return;
  int k = idx / Nc, n = idx % Nc;
  float x = (k < Ksrc) ? srcw[(size_t)k * Nc + n] : 0.0f;
  dstw[((size_t)(k >> 4) * Nc + n) * 16 + (k & 15)] = f2bf(x);
}

// ---------------- elementwise kernels ----------------
__global__ void k_s_init(const int* __restrict__ an, const float* __restrict__ embed,
                         float* __restrict__ s) {
  int i = blockIdx.x * 256 + threadIdx.x;
  if (i >= NN * HH) return;
  int n = i / HH, h = i % HH;
  s[i] = embed[(size_t)an[n] * HH + h];
}

__global__ void k_edge_geom(const float* __restrict__ pos, const int* __restrict__ src,
                            const int* __restrict__ dst, float* __restrict__ dvn,
                            float* __restrict__ rbf) {
  int e = blockIdx.x * 256 + threadIdx.x;
  if (e >= EE) return;
  int a = src[e], b = dst[e];
  float dx = pos[a * 3 + 0] - pos[b * 3 + 0];
  float dy = pos[a * 3 + 1] - pos[b * 3 + 1];
  float dz = pos[a * 3 + 2] - pos[b * 3 + 2];
  float dist = sqrtf(dx * dx + dy * dy + dz * dz);
  float inv = 1.0f / (dist + 1e-8f);
  dvn[e * 3 + 0] = dx * inv; dvn[e * 3 + 1] = dy * inv; dvn[e * 3 + 2] = dz * inv;
  const float step = CUT / (float)(GG - 1);
  const float coeff = -0.5f / (step * step);
  float* r = rbf + (size_t)e * GP;
  #pragma unroll
  for (int g = 0; g < GP; ++g) {
    float d = dist - step * (float)g;
    r[g] = (g < GG) ? expf(coeff * d * d) : 0.0f;
  }
}

__global__ void k_vnorm(const float* __restrict__ vt, float* __restrict__ vn) {
  int i = blockIdx.x * 256 + threadIdx.x;
  if (i >= NN * HH) return;
  int n = i / HH, h = i % HH;
  float acc = 0.0f;
  #pragma unroll
  for (int d = 0; d < 3; ++d) {
    float x = vt[((size_t)n * 3 + d) * H2 + h];   // V1 = vt[..., :H]
    acc += x * x;
  }
  vn[i] = sqrtf(acc);
}

__global__ void k_update(const float* __restrict__ u, const float* __restrict__ vt,
                         float* __restrict__ s, float* __restrict__ v) {
  int i = blockIdx.x * 256 + threadIdx.x;
  if (i >= NN * HH) return;
  int n = i / HH, h = i % HH;
  const float* un = u + (size_t)n * H3;
  float ua = un[h], ub = un[HH + h], uc = un[2 * HH + h];
  s[i] += ua;
  #pragma unroll
  for (int d = 0; d < 3; ++d) {
    size_t vi = ((size_t)n * 3 + d) * HH + h;
    size_t ti = ((size_t)n * 3 + d) * H2 + HH + h;  // V2 = vt[..., H:2H]
    v[vi] = v[vi] * ub + vt[ti] * uc;
  }
}

__global__ void k_readout(const float* __restrict__ y1, const float* __restrict__ ro_w2,
                          const float* __restrict__ ro_b2, const int* __restrict__ bids,
                          float* __restrict__ sums, float* __restrict__ counts) {
  int n = blockIdx.x * 256 + threadIdx.x;
  if (n >= NN) return;
  float acc = ro_b2[0];
  #pragma unroll
  for (int j = 0; j < 64; ++j) acc += y1[(size_t)n * 64 + j] * ro_w2[j];
  int g = bids[n];
  atomicAdd(&sums[g], acc);
  atomicAdd(&counts[g], 1.0f);
}

__global__ void k_final(const float* __restrict__ sums, const float* __restrict__ counts,
                        float* __restrict__ out) {
  int g = threadIdx.x;
  if (g < NGR) out[g] = sums[g] / fmaxf(counts[g], 1.0f);
}

// ---------------- host ----------------
extern "C" void kernel_launch(void* const* d_in, const int* in_sizes, int n_in,
                              void* d_out, int out_size, void* d_ws, size_t ws_size,
                              hipStream_t stream) {
  (void)in_sizes; (void)n_in; (void)out_size; (void)ws_size;
  const int*   an     = (const int*)d_in[0];
  const float* pos    = (const float*)d_in[1];
  const int*   eidx   = (const int*)d_in[2];
  const int*   bids   = (const int*)d_in[3];
  const float* embed  = (const float*)d_in[4];
  const float* msg_w1 = (const float*)d_in[5];
  const float* msg_b1 = (const float*)d_in[6];
  const float* msg_w2 = (const float*)d_in[7];
  const float* msg_b2 = (const float*)d_in[8];
  const float* filt_w = (const float*)d_in[9];
  const float* filt_b = (const float*)d_in[10];
  const float* vec_w  = (const float*)d_in[11];
  const float* upd_w1 = (const float*)d_in[12];
  const float* upd_b1 = (const float*)d_in[13];
  const float* upd_w2 = (const float*)d_in[14];
  const float* upd_b2 = (const float*)d_in[15];
  const float* ro_w1  = (const float*)d_in[16];
  const float* ro_b1  = (const float*)d_in[17];
  const float* ro_w2  = (const float*)d_in[18];
  const float* ro_b2  = (const float*)d_in[19];
  const int* src = eidx;
  const int* dst = eidx + EE;
  float* out = (float*)d_out;

  char* base = (char*)d_ws;
  size_t off = 0;
  auto alloc_f = [&](size_t n) -> float* {
    float* p = (float*)(base + off);
    off += ((n * 4 + 255) & ~(size_t)255);
    return p;
  };
  auto alloc_u16 = [&](size_t n) -> unsigned short* {
    unsigned short* p = (unsigned short*)(base + off);
    off += ((n * 2 + 255) & ~(size_t)255);
    return p;
  };
  float* s_buf  = alloc_f((size_t)NN * HH);
  float* v_buf  = alloc_f((size_t)NN * 3 * HH);
  float* vcopy  = alloc_f((size_t)NN * 3 * HH);
  float* sh_buf = alloc_f((size_t)NN * HH);      // s_h / u1 / y1 (reused)
  float* sall   = alloc_f((size_t)NN * H3);      // s_all / u (reused)
  float* vt_buf = alloc_f((size_t)NN * 3 * H2);
  float* vn_buf = alloc_f((size_t)NN * HH);
  float* rbf    = alloc_f((size_t)EE * GP);
  float* dvn    = alloc_f((size_t)EE * 3);
  float* sums   = alloc_f(NGR);
  float* counts = alloc_f(NGR);
  float* zbias  = alloc_f(H3);                   // zero bias for bias-less GEMMs
  unsigned short* pk_filt = alloc_u16((size_t)LL * 4 * H3 * 16);
  unsigned short* pk_m1   = alloc_u16((size_t)LL * 8 * HH * 16);
  unsigned short* pk_m2   = alloc_u16((size_t)LL * 8 * H3 * 16);
  unsigned short* pk_vec  = alloc_u16((size_t)LL * 8 * H2 * 16);
  unsigned short* pk_u1   = alloc_u16((size_t)LL * 16 * HH * 16);
  unsigned short* pk_u2   = alloc_u16((size_t)LL * 8 * H3 * 16);
  unsigned short* pk_ro1  = alloc_u16((size_t)8 * 64 * 16);

  (void)hipMemsetAsync(v_buf, 0, (size_t)NN * 3 * HH * 4, stream);
  (void)hipMemsetAsync(sums, 0, NGR * 4, stream);
  (void)hipMemsetAsync(counts, 0, NGR * 4, stream);
  (void)hipMemsetAsync(zbias, 0, H3 * 4, stream);

  auto pack = [&](const float* sw, unsigned short* dw, int Ksrc, int Kpad, int Nc) {
    int tot = Kpad * Nc;
    pack_b_kernel<<<(tot + 255) / 256, 256, 0, stream>>>(sw, dw, Ksrc, Kpad, Nc);
  };
  for (int i = 0; i < LL; ++i) {
    pack(filt_w + (size_t)i * GG * H3, pk_filt + (size_t)i * 4 * H3 * 16, GG, GP, H3);
    pack(msg_w1 + (size_t)i * HH * HH, pk_m1 + (size_t)i * 8 * HH * 16, HH, HH, HH);
    pack(msg_w2 + (size_t)i * HH * H3, pk_m2 + (size_t)i * 8 * H3 * 16, HH, HH, H3);
    pack(vec_w  + (size_t)i * HH * H2, pk_vec + (size_t)i * 8 * H2 * 16, HH, HH, H2);
    pack(upd_w1 + (size_t)i * H2 * HH, pk_u1 + (size_t)i * 16 * HH * 16, H2, H2, HH);
    pack(upd_w2 + (size_t)i * HH * H3, pk_u2 + (size_t)i * 8 * H3 * 16, HH, HH, H3);
  }
  pack(ro_w1, pk_ro1, HH, HH, 64);

  k_s_init<<<(NN * HH + 255) / 256, 256, 0, stream>>>(an, embed, s_buf);
  k_edge_geom<<<(EE + 255) / 256, 256, 0, stream>>>(pos, src, dst, dvn, rbf);

  auto gemm = [&](const float* A1, int K1, const float* A2, int K2,
                  const unsigned short* Bp, const float* bias, float* o,
                  int M, int Nld, int relu) {
    dim3 g((M + 127) / 128, Nld / 64);
    gemm_bf16_kernel<<<g, 256, 0, stream>>>(A1, K1, A2, K2, Bp, bias, o, M, Nld, relu);
  };

  for (int i = 0; i < LL; ++i) {
    // message block
    gemm(s_buf, HH, (const float*)nullptr, 0, pk_m1 + (size_t)i * 8 * HH * 16,
         msg_b1 + (size_t)i * HH, sh_buf, NN, HH, 1);
    gemm(sh_buf, HH, (const float*)nullptr, 0, pk_m2 + (size_t)i * 8 * H3 * 16,
         msg_b2 + (size_t)i * H3, sall, NN, H3, 0);
    (void)hipMemcpyAsync(vcopy, v_buf, (size_t)NN * 3 * HH * 4, hipMemcpyDeviceToDevice, stream);
    k_filt_gate_scatter<<<dim3(EE / 128, 6), 256, 0, stream>>>(
        rbf, pk_filt + (size_t)i * 4 * H3 * 16, filt_b + (size_t)i * H3,
        sall, vcopy, dvn, src, dst, s_buf, v_buf);
    // update block
    gemm(v_buf, HH, (const float*)nullptr, 0, pk_vec + (size_t)i * 8 * H2 * 16,
         zbias, vt_buf, NN * 3, H2, 0);
    k_vnorm<<<(NN * HH + 255) / 256, 256, 0, stream>>>(vt_buf, vn_buf);
    gemm(s_buf, HH, vn_buf, HH, pk_u1 + (size_t)i * 16 * HH * 16,
         upd_b1 + (size_t)i * HH, sh_buf, NN, HH, 1);
    gemm(sh_buf, HH, (const float*)nullptr, 0, pk_u2 + (size_t)i * 8 * H3 * 16,
         upd_b2 + (size_t)i * H3, sall, NN, H3, 0);
    k_update<<<(NN * HH + 255) / 256, 256, 0, stream>>>(sall, vt_buf, s_buf, v_buf);
  }

  // readout + global mean pool
  gemm(s_buf, HH, (const float*)nullptr, 0, pk_ro1, ro_b1, sh_buf, NN, 64, 1);
  k_readout<<<(NN + 255) / 256, 256, 0, stream>>>(sh_buf, ro_w2, ro_b2, bids, sums, counts);
  k_final<<<1, NGR, 0, stream>>>(sums, counts, out);
}
